// QLSTMGen_65481071401314
// MI455X (gfx1250) — matrix operand order repaired and checked by reading, not compiled, not run on hardware
//
#include <hip/hip_runtime.h>
#include <cstddef>

// ---------------------------------------------------------------------------
// CDNA5 (gfx1250) bf16 WMMA implementation of the LSTM->phase->fe->GRU stack.
// wave32; V_WMMA_F32_16X16X32_BF16 with f32 accumulation.
// ---------------------------------------------------------------------------

typedef __attribute__((ext_vector_type(16))) __bf16        v16bf;
typedef __attribute__((ext_vector_type(8)))  float         v8f;
typedef __attribute__((ext_vector_type(4)))  unsigned int  uv4;
typedef __attribute__((ext_vector_type(2)))  unsigned int  uv2;

union FragAB { v16bf v; uv4 q[2]; };
union Pack4  { __bf16 h[4]; uv2 u; };

constexpr int TT = 512;        // time steps
constexpr int BB = 256;        // batch
constexpr int DD = 256;        // input dim
constexpr int HH = 512;        // hidden
constexpr int G4 = 4 * HH;     // 2048 LSTM gate rows
constexpr int KX = DD + HH;    // 768 fused LSTM reduction

__device__ __forceinline__ float sigf(float x) { return 1.0f / (1.0f + __expf(-x)); }

__device__ __forceinline__ v8f v8f_zero() {
  v8f z;
#pragma unroll
  for (int i = 0; i < 8; ++i) z[i] = 0.0f;
  return z;
}

__device__ __forceinline__ v8f wmma_bf16(const FragAB& a, const FragAB& b, v8f c) {
  // D = A(16x32 bf16) * B(32x16 bf16) + C(16x16 f32)
  return __builtin_amdgcn_wmma_f32_16x16x32_bf16(false, a.v, false, b.v,
                                                 (short)0, c, false, false);
}

// ---------------------------------------------------------------------------
// GEMM micro-kernels.  A: LDS 16x512 bf16 row-major (K contiguous).
// Bw: global [N][512] bf16, row-major in K (pre-transposed), so each lane's
// 16 K-values are two b128 loads at immediate offsets from a hoisted row
// pointer.  All B fragments are loaded into distinct registers before the
// WMMA burst so the WAR hazard slots overlap with the next loads.
// ---------------------------------------------------------------------------
__device__ __forceinline__ void gemm4_k512(const __bf16* As, const __bf16* __restrict__ Bw,
                                           int colBase, int lane, v8f acc[4]) {
  const int m     = lane & 15;
  const int koff  = (lane & 16) ? 16 : 0;   // A-matrix K sub-block per lane-half
  const int kboff = (lane & 16) ? 16 : 0;   // B-matrix K sub-block per lane-half
  const __bf16* ap = As + m * 512 + koff;
  const __bf16* bp0 = Bw + (size_t)(colBase +  0 + m) * 512 + kboff;
  const __bf16* bp1 = Bw + (size_t)(colBase + 16 + m) * 512 + kboff;
  const __bf16* bp2 = Bw + (size_t)(colBase + 32 + m) * 512 + kboff;
  const __bf16* bp3 = Bw + (size_t)(colBase + 48 + m) * 512 + kboff;
  __builtin_prefetch(bp0, 0, 1);
  __builtin_prefetch(bp1, 0, 1);
  __builtin_prefetch(bp2, 0, 1);
  __builtin_prefetch(bp3, 0, 1);
#pragma unroll 4
  for (int kb = 0; kb < 512; kb += 32) {
    FragAB a, b0, b1, b2, b3;
    a.q[0]  = *(const uv4*)(ap  + kb);
    a.q[1]  = *(const uv4*)(ap  + kb + 8);
    b0.q[0] = *(const uv4*)(bp0 + kb);  b0.q[1] = *(const uv4*)(bp0 + kb + 8);
    b1.q[0] = *(const uv4*)(bp1 + kb);  b1.q[1] = *(const uv4*)(bp1 + kb + 8);
    b2.q[0] = *(const uv4*)(bp2 + kb);  b2.q[1] = *(const uv4*)(bp2 + kb + 8);
    b3.q[0] = *(const uv4*)(bp3 + kb);  b3.q[1] = *(const uv4*)(bp3 + kb + 8);
    acc[0] = wmma_bf16(a, b0, acc[0]);
    acc[1] = wmma_bf16(a, b1, acc[1]);
    acc[2] = wmma_bf16(a, b2, acc[2]);
    acc[3] = wmma_bf16(a, b3, acc[3]);
  }
}

// Three 16-wide tiles at columns j0, HH+j0, 2*HH+j0 sharing one A fragment
// per K-chunk (GRU r/z/n hidden GEMM).
__device__ __forceinline__ void gemm3_k512(const __bf16* As, const __bf16* __restrict__ Bw,
                                           int j0, int lane, v8f acc[3]) {
  const int m     = lane & 15;
  const int koff  = (lane & 16) ? 16 : 0;
  const int kboff = (lane & 16) ? 16 : 0;
  const __bf16* ap = As + m * 512 + koff;
  const __bf16* bp0 = Bw + (size_t)(j0 +          m) * 512 + kboff;
  const __bf16* bp1 = Bw + (size_t)(j0 + HH     + m) * 512 + kboff;
  const __bf16* bp2 = Bw + (size_t)(j0 + 2 * HH + m) * 512 + kboff;
  __builtin_prefetch(bp0, 0, 1);
  __builtin_prefetch(bp1, 0, 1);
  __builtin_prefetch(bp2, 0, 1);
#pragma unroll 4
  for (int kb = 0; kb < 512; kb += 32) {
    FragAB a, b0, b1, b2;
    a.q[0]  = *(const uv4*)(ap  + kb);
    a.q[1]  = *(const uv4*)(ap  + kb + 8);
    b0.q[0] = *(const uv4*)(bp0 + kb);  b0.q[1] = *(const uv4*)(bp0 + kb + 8);
    b1.q[0] = *(const uv4*)(bp1 + kb);  b1.q[1] = *(const uv4*)(bp1 + kb + 8);
    b2.q[0] = *(const uv4*)(bp2 + kb);  b2.q[1] = *(const uv4*)(bp2 + kb + 8);
    acc[0] = wmma_bf16(a, b0, acc[0]);
    acc[1] = wmma_bf16(a, b1, acc[1]);
    acc[2] = wmma_bf16(a, b2, acc[2]);
  }
}

// One phase of the fused LSTM GEMM: acc += A(KD slice) @ W^T.
// src rows have stride == KD (x for KD=256, h for KD=512).  A tile is
// double-buffered in LDS with a single barrier per K-chunk: each wave's
// previous-buffer reads are DScnt-complete before it signals the barrier.
template <int KD>
__device__ __forceinline__ void lstm_gemm_phase(
    const float* __restrict__ src, int m0,
    const __bf16* __restrict__ W, int gcol,
    __bf16* AldBase, int& sel, int tid, int lane, v8f acc[4])
{
  const int m     = lane & 15;
  const int koff  = (lane & 16) ? 16 : 0;
  const int kboff = (lane & 16) ? 16 : 0;
  const __bf16* bp0 = W + (size_t)(gcol +  0 + m) * KD + kboff;
  const __bf16* bp1 = W + (size_t)(gcol + 16 + m) * KD + kboff;
  const __bf16* bp2 = W + (size_t)(gcol + 32 + m) * KD + kboff;
  const __bf16* bp3 = W + (size_t)(gcol + 48 + m) * KD + kboff;
  __builtin_prefetch(bp0, 0, 1);
  __builtin_prefetch(bp1, 0, 1);
  __builtin_prefetch(bp2, 0, 1);
  __builtin_prefetch(bp3, 0, 1);
  const int mmS = tid >> 3;            // staging: 128 thr x 1 float4 = 512 elems
  const int k4S = (tid & 7) * 4;
  const float* srcRow = src + (size_t)(m0 + mmS) * KD + k4S;

  for (int kb = 0; kb < KD; kb += 32) {
    __bf16* buf = AldBase + (sel & 1) * (16 * 32);
    sel ^= 1;
    {
      const float4 f = *(const float4*)(srcRow + kb);
      Pack4 p;
      p.h[0] = (__bf16)f.x; p.h[1] = (__bf16)f.y;
      p.h[2] = (__bf16)f.z; p.h[3] = (__bf16)f.w;
      *(uv2*)(buf + mmS * 32 + k4S) = p.u;
    }
    __syncthreads();
    FragAB a, b0, b1, b2, b3;
    a.q[0]  = *(const uv4*)(buf + m * 32 + koff);
    a.q[1]  = *(const uv4*)(buf + m * 32 + 8 + koff);
    b0.q[0] = *(const uv4*)(bp0 + kb);  b0.q[1] = *(const uv4*)(bp0 + kb + 8);
    b1.q[0] = *(const uv4*)(bp1 + kb);  b1.q[1] = *(const uv4*)(bp1 + kb + 8);
    b2.q[0] = *(const uv4*)(bp2 + kb);  b2.q[1] = *(const uv4*)(bp2 + kb + 8);
    b3.q[0] = *(const uv4*)(bp3 + kb);  b3.q[1] = *(const uv4*)(bp3 + kb + 8);
    acc[0] = wmma_bf16(a, b0, acc[0]);
    acc[1] = wmma_bf16(a, b1, acc[1]);
    acc[2] = wmma_bf16(a, b2, acc[2]);
    acc[3] = wmma_bf16(a, b3, acc[3]);
  }
}

// ---------------------------------------------------------------------------
// Kernel A: fused LSTM step.  grid (B/16, H/64), 128 threads = 4 waves,
// wave g computes gate g (i/f/g/o order) so the nonlinearity fuses in-block.
// gates = x@Wx^T + h@Wh_sum^T + bias ; c' = f*c + i*g ; h_lstm = o*tanh(c')
// ---------------------------------------------------------------------------
__global__ __launch_bounds__(128) void qlstm_lstm_step(
    const float* __restrict__ x_all, const __bf16* __restrict__ wx,
    const __bf16* __restrict__ wh, const float* __restrict__ biasl,
    const float* __restrict__ hprev, float* __restrict__ cx,
    float* __restrict__ hl, int t)
{
  __shared__ __bf16 Ald[2][16 * 32];  // double-buffered staged A tile
  __shared__ float  Gs[4 * 16 * 64];  // gate pre-activations for pointwise

  const int tid  = threadIdx.x;
  const int g    = tid >> 5;          // wave id == gate id
  const int lane = tid & 31;
  const int m0   = blockIdx.x * 16;   // batch-row base
  const int c0   = blockIdx.y * 64;   // column base within a gate
  const int m    = lane & 15;
  const float* xt = x_all + (size_t)t * BB * DD;
  const int gcol = g * HH + c0;

  v8f acc[4];
#pragma unroll
  for (int nt = 0; nt < 4; ++nt) acc[nt] = v8f_zero();

  int sel = 0;
  lstm_gemm_phase<DD>(xt,    m0, wx, gcol, &Ald[0][0], sel, tid, lane, acc);
  lstm_gemm_phase<HH>(hprev, m0, wh, gcol, &Ald[0][0], sel, tid, lane, acc);

  // spill gate pre-activations (+fused bias) for cross-gate pointwise
#pragma unroll
  for (int nt = 0; nt < 4; ++nt)
#pragma unroll
    for (int r = 0; r < 8; ++r) {
      int mm = r + ((lane & 16) ? 8 : 0);
      int cc = nt * 16 + m;
      Gs[(g * 16 + mm) * 64 + cc] = acc[nt][r] + biasl[gcol + cc];
    }
  __syncthreads();

#pragma unroll
  for (int i = 0; i < 8; ++i) {       // 128 thr x 8 = 1024 cells
    int s  = tid * 8 + i;
    int mm = s >> 6, cc = s & 63;
    float iv = Gs[(0 * 16 + mm) * 64 + cc];
    float fv = Gs[(1 * 16 + mm) * 64 + cc];
    float gv = Gs[(2 * 16 + mm) * 64 + cc];
    float ov = Gs[(3 * 16 + mm) * 64 + cc];
    float ig = sigf(iv), fg = sigf(fv), gg = tanhf(gv), og = sigf(ov);
    size_t idx = (size_t)(m0 + mm) * HH + (c0 + cc);
    float c = fg * cx[idx] + ig * gg;
    cx[idx] = c;
    hl[idx] = og * tanhf(c);
  }
}

// ---------------------------------------------------------------------------
// Kernel B: fused phase -> fe -> GRU.  One block owns a 16-row batch stripe so
// every chained GEMM's full K=512 row is in LDS.  8 waves x 64 cols = 512.
// Dynamic LDS: hA(16KB) + [hB(16KB) overlaid by GI f32 96KB] = 112KB (<320KB/WGP)
// ---------------------------------------------------------------------------
__global__ __launch_bounds__(256) void qlstm_tail_step(
    const float* __restrict__ hl, const __bf16* __restrict__ phT,
    const __bf16* __restrict__ feW, const float* __restrict__ feB,
    const __bf16* __restrict__ gWih, const __bf16* __restrict__ gWhh,
    const float* __restrict__ gbih, const float* __restrict__ gbhh,
    float* __restrict__ hstate, float* __restrict__ out, int t)
{
  extern __shared__ char smemB[];
  __bf16* hA = (__bf16*)smemB;             // 16x512 activation (bf16)
  __bf16* hB = (__bf16*)(smemB + 16384);   // 16x512, dead after fe GEMM
  float*  GI = (float*)(smemB + 16384);    // 16x1536 gi pre-acts (overlays hB)

  const int tid  = threadIdx.x;
  const int w    = tid >> 5;               // 8 waves
  const int lane = tid & 31;
  const int m0   = blockIdx.x * 16;
  const int m    = lane & 15;
  const int hi8  = (lane & 16) ? 8 : 0;

  // stage h_lstm stripe (contiguous 16x512 region), f32 -> packed bf16
  {
    const float4* src = (const float4*)(hl + (size_t)m0 * HH);
#pragma unroll
    for (int i = 0; i < 8; ++i) {
      int idx = i * 256 + tid;             // 0..2047 float4s
      const float4 f = src[idx];
      Pack4 p;
      p.h[0] = (__bf16)f.x; p.h[1] = (__bf16)f.y;
      p.h[2] = (__bf16)f.z; p.h[3] = (__bf16)f.w;
      *(uv2*)(hA + idx * 4) = p.u;
    }
  }
  __syncthreads();

  // P1: h2 = h_lstm @ phase   (phT is phase pre-transposed, K contiguous)
  {
    v8f acc[4];
#pragma unroll
    for (int nt = 0; nt < 4; ++nt) acc[nt] = v8f_zero();
    gemm4_k512(hA, phT, w * 64, lane, acc);
#pragma unroll
    for (int nt = 0; nt < 4; ++nt)
#pragma unroll
      for (int r = 0; r < 8; ++r)
        hB[(r + hi8) * 512 + w * 64 + nt * 16 + m] = (__bf16)acc[nt][r];
  }
  __syncthreads();

  // P2: h3 = h2 @ fe_W^T + fe_b   -> back into hA
  {
    v8f acc[4];
#pragma unroll
    for (int nt = 0; nt < 4; ++nt) acc[nt] = v8f_zero();
    gemm4_k512(hB, feW, w * 64, lane, acc);
#pragma unroll
    for (int nt = 0; nt < 4; ++nt)
#pragma unroll
      for (int r = 0; r < 8; ++r) {
        int col = w * 64 + nt * 16 + m;
        hA[(r + hi8) * 512 + col] = (__bf16)(acc[nt][r] + feB[col]);
      }
  }
  __syncthreads();

  // P3: gi = h3 @ gru_Wih^T + gru_bih  (this wave's 64 j-cols of r/z/n)
  for (int g = 0; g < 3; ++g) {
    v8f acc[4];
#pragma unroll
    for (int nt = 0; nt < 4; ++nt) acc[nt] = v8f_zero();
    gemm4_k512(hA, gWih, g * HH + w * 64, lane, acc);
#pragma unroll
    for (int nt = 0; nt < 4; ++nt)
#pragma unroll
      for (int r = 0; r < 8; ++r) {
        int col = g * HH + w * 64 + nt * 16 + m;
        GI[(r + hi8) * 1536 + col] = acc[nt][r] + gbih[col];
      }
  }
  // no barrier: each wave reads back only its own GI columns (same-wave LDS
  // ops are kept in order by DScnt)

  // P4: gh = h3 @ gru_Whh^T + gru_bhh, then GRU combine per 16-wide j tile.
  // gemm3 shares one A fragment across the r/z/n column tiles.
  for (int nt = 0; nt < 4; ++nt) {
    int j0 = w * 64 + nt * 16;
    v8f gh[3];
#pragma unroll
    for (int q = 0; q < 3; ++q) gh[q] = v8f_zero();
    gemm3_k512(hA, gWhh, j0, lane, gh);
#pragma unroll
    for (int r = 0; r < 8; ++r) {
      int mm = r + hi8;
      int j  = j0 + m;
      float hr = gh[0][r] + gbhh[j];
      float hz = gh[1][r] + gbhh[HH + j];
      float hn = gh[2][r] + gbhh[2 * HH + j];
      float ir  = GI[mm * 1536 + j];
      float iz  = GI[mm * 1536 + HH + j];
      float inn = GI[mm * 1536 + 2 * HH + j];
      float rg = sigf(ir + hr);
      float zg = sigf(iz + hz);
      float ng = tanhf(inn + rg * hn);
      float h3v = (float)hA[mm * 512 + j];
      float hnew = (1.0f - zg) * ng + zg * h3v;
      size_t gidx = (size_t)(m0 + mm) * HH + j;
      out[(size_t)t * BB * HH + gidx] = hnew;
      hstate[gidx] = hnew;
    }
  }
}

// ---------------------------------------------------------------------------
// One-time (per launch) weight prep: bf16 conversion, transposes, bias fusion.
// ---------------------------------------------------------------------------
__global__ void k_prep_wx(const float* __restrict__ Wih, __bf16* __restrict__ wx) {
  int i = blockIdx.x * blockDim.x + threadIdx.x;
  if (i < G4 * DD) { int r = i >> 8, c = i & 255; wx[i] = (__bf16)Wih[r * KX + c]; }
}
__global__ void k_prep_wh(const float* __restrict__ Wih, const float* __restrict__ Whh,
                          __bf16* __restrict__ wh) {
  int i = blockIdx.x * blockDim.x + threadIdx.x;
  if (i < G4 * HH) {
    int r = i >> 9, c = i & 511;
    wh[i] = (__bf16)(Wih[r * KX + DD + c] + Whh[i]);
  }
}
__global__ void k_addf(const float* a, const float* b, float* d, int n) {
  int i = blockIdx.x * blockDim.x + threadIdx.x;
  if (i < n) d[i] = a[i] + b[i];
}
__global__ void k_prep_phT(const float* __restrict__ phase, __bf16* __restrict__ phT) {
  int i = blockIdx.x * blockDim.x + threadIdx.x;
  if (i < HH * HH) { int nn = i >> 9, k = i & 511; phT[i] = (__bf16)phase[k * HH + nn]; }
}
__global__ void k_cvt(const float* __restrict__ s, __bf16* __restrict__ d, int n) {
  int i = blockIdx.x * blockDim.x + threadIdx.x;
  if (i < n) d[i] = (__bf16)s[i];
}
__global__ void k_zerof(float* p, int n) {
  int i = blockIdx.x * blockDim.x + threadIdx.x;
  if (i < n) p[i] = 0.0f;
}

// ---------------------------------------------------------------------------
extern "C" void kernel_launch(void* const* d_in, const int* in_sizes, int n_in,
                              void* d_out, int out_size, void* d_ws, size_t ws_size,
                              hipStream_t stream) {
  (void)in_sizes; (void)n_in; (void)out_size; (void)ws_size;
  const float* inputs = (const float*)d_in[0];
  const float* lWih   = (const float*)d_in[1];
  const float* lWhh   = (const float*)d_in[2];
  const float* lbih   = (const float*)d_in[3];
  const float* lbhh   = (const float*)d_in[4];
  const float* phase  = (const float*)d_in[5];
  const float* feW    = (const float*)d_in[6];
  const float* feB    = (const float*)d_in[7];
  const float* gWih   = (const float*)d_in[8];
  const float* gWhh   = (const float*)d_in[9];
  const float* gbih   = (const float*)d_in[10];
  const float* gbhh   = (const float*)d_in[11];

  char* ws = (char*)d_ws;
  size_t off = 0;
  auto alloc = [&](size_t bytes) -> char* {
    char* p = ws + off;
    off = (off + bytes + 255) & ~(size_t)255;
    return p;
  };
  __bf16* wx     = (__bf16*)alloc((size_t)G4 * DD * 2);      // 1 MB
  __bf16* wh     = (__bf16*)alloc((size_t)G4 * HH * 2);      // 2 MB
  float*  biasl  = (float*) alloc((size_t)G4 * 4);           // 8 KB
  __bf16* phT    = (__bf16*)alloc((size_t)HH * HH * 2);      // 512 KB
  __bf16* feWb   = (__bf16*)alloc((size_t)HH * HH * 2);      // 512 KB
  __bf16* gWihb  = (__bf16*)alloc((size_t)3 * HH * HH * 2);  // 1.5 MB
  __bf16* gWhhb  = (__bf16*)alloc((size_t)3 * HH * HH * 2);  // 1.5 MB
  float*  hstate = (float*) alloc((size_t)BB * HH * 4);      // 512 KB
  float*  cx     = (float*) alloc((size_t)BB * HH * 4);      // 512 KB
  float*  hlin   = (float*) alloc((size_t)BB * HH * 4);      // 512 KB

  int n;
  n = G4 * DD;     k_prep_wx <<<(n + 255) / 256, 256, 0, stream>>>(lWih, wx);
  n = G4 * HH;     k_prep_wh <<<(n + 255) / 256, 256, 0, stream>>>(lWih, lWhh, wh);
  n = G4;          k_addf    <<<(n + 255) / 256, 256, 0, stream>>>(lbih, lbhh, biasl, n);
  n = HH * HH;     k_prep_phT<<<(n + 255) / 256, 256, 0, stream>>>(phase, phT);
  n = HH * HH;     k_cvt     <<<(n + 255) / 256, 256, 0, stream>>>(feW, feWb, n);
  n = 3 * HH * HH; k_cvt     <<<(n + 255) / 256, 256, 0, stream>>>(gWih, gWihb, n);
  n = 3 * HH * HH; k_cvt     <<<(n + 255) / 256, 256, 0, stream>>>(gWhh, gWhhb, n);
  n = BB * HH;     k_zerof   <<<(n + 255) / 256, 256, 0, stream>>>(hstate, n);
  n = BB * HH;     k_zerof   <<<(n + 255) / 256, 256, 0, stream>>>(cx, n);

  float* out = (float*)d_out;
  const size_t smemB = 16384 + (size_t)16 * 1536 * 4;  // 112 KB dynamic LDS
  for (int t = 0; t < TT; ++t) {
    qlstm_lstm_step<<<dim3(BB / 16, HH / 64), 128, 0, stream>>>(
        inputs, wx, wh, biasl, hstate, cx, hlin, t);
    qlstm_tail_step<<<dim3(BB / 16), 256, smemB, stream>>>(
        hlin, phT, feWb, feB, gWihb, gWhhb, gbih, gbhh, hstate, out, t);
  }

  // final hx = outputs[T-1], cx from workspace
  size_t outsElems = (size_t)TT * BB * HH;
  hipMemcpyAsync(out + outsElems, out + outsElems - (size_t)BB * HH,
                 (size_t)BB * HH * sizeof(float), hipMemcpyDeviceToDevice, stream);
  hipMemcpyAsync(out + outsElems + (size_t)BB * HH, cx,
                 (size_t)BB * HH * sizeof(float), hipMemcpyDeviceToDevice, stream);
}
